// CntkFeatureMapOps_9414568313531
// MI455X (gfx1250) — matrix operand order repaired and checked by reading, not compiled, not run on hardware
//
#include <hip/hip_runtime.h>
#include <hip/hip_bf16.h>
#include <stdint.h>

// ---------------------------------------------------------------- constants
#define NROWS  4096
#define DNNGP  8192
#define MDIM   2048      // M0 == M1
#define MCS    4096
#define DNTK   24576
#define CATW   (MDIM + MCS)   // 6144

typedef __attribute__((ext_vector_type(16))) __bf16 v16bf;
typedef __attribute__((ext_vector_type(8)))  __bf16 v8bf;
typedef __attribute__((ext_vector_type(8)))  float  v8f;
typedef __attribute__((ext_vector_type(4)))  float  v4f;

// float -> bf16, round to nearest even
__device__ __forceinline__ __bf16 f2bf(float f) {
    union { float f; uint32_t u; } cv; cv.f = f;
    uint32_t u = cv.u;
    u += 0x7FFFu + ((u >> 16) & 1u);
    union { unsigned short s; __bf16 b; } r; r.s = (unsigned short)(u >> 16);
    return r.b;
}

__device__ __forceinline__ v16bf cat8(v8bf lo, v8bf hi) {
    return __builtin_shufflevector(lo, hi, 0,1,2,3,4,5,6,7,8,9,10,11,12,13,14,15);
}

// gfx1250 async global->LDS copy, 16 bytes per active lane (ASYNCcnt tracked)
__device__ __forceinline__ void async_copy_b128(unsigned int lds_addr,
                                                unsigned long long gaddr) {
    asm volatile("global_load_async_to_lds_b128 %0, %1, off"
                 :: "v"(lds_addr), "v"(gaddr) : "memory");
}

__device__ __forceinline__ void wait_async0() {
#if __has_builtin(__builtin_amdgcn_s_wait_asynccnt)
    __builtin_amdgcn_s_wait_asynccnt(0);
#else
    asm volatile("s_wait_asynccnt 0x0" ::: "memory");
#endif
}

// ------------------------------------------------- prep: f32 -> bf16 convert
__global__ __launch_bounds__(256) void convert_bf16_kernel(
        const float* __restrict__ src, __bf16* __restrict__ dst) {
    size_t i = ((size_t)blockIdx.x * 256 + threadIdx.x) * 4;
    v4f f = *(const v4f*)(src + i);
    dst[i+0] = f2bf(f[0]); dst[i+1] = f2bf(f[1]);
    dst[i+2] = f2bf(f[2]); dst[i+3] = f2bf(f[3]);
}

// --------------------------- prep: W [8192 k][2048 n] f32 -> Wt [n][k] bf16
__global__ __launch_bounds__(256) void transpose_bf16_kernel(
        const float* __restrict__ W, __bf16* __restrict__ Wt) {
    __shared__ __bf16 tile[64][65];           // [n][k], padded
    const int kb  = blockIdx.x * 64;
    const int nb  = blockIdx.y * 64;
    const int tid = threadIdx.x;
    #pragma unroll
    for (int i = 0; i < 4; ++i) {             // 1024 float4 reads
        int idx = tid + i * 256;
        int k   = idx >> 4;                   // 16 float4 per k-row
        int n4  = (idx & 15) * 4;
        v4f f = *(const v4f*)(W + (size_t)(kb + k) * MDIM + nb + n4);
        tile[n4+0][k] = f2bf(f[0]); tile[n4+1][k] = f2bf(f[1]);
        tile[n4+2][k] = f2bf(f[2]); tile[n4+3][k] = f2bf(f[3]);
    }
    __syncthreads();
    #pragma unroll
    for (int i = 0; i < 4; ++i) {             // 1024 x 4-elem writes
        int idx = tid + i * 256;
        int n   = idx >> 4;
        int k4  = (idx & 15) * 4;
        __bf16* o = Wt + (size_t)(nb + n) * DNNGP + kb + k4;
        o[0] = tile[n][k4+0]; o[1] = tile[n][k4+1];
        o[2] = tile[n][k4+2]; o[3] = tile[n][k4+3];
    }
}

// ---------------------------------------------------------------- WMMA GEMM
// C[4096 x 2048] = A[4096 x 8192]bf16 * Bt[2048 n][8192 k]bf16, f32 accum.
// Double-buffered LDS filled by async global->LDS b128 copies.
// MODE 0: (c > 0) * norm0 -> out0.   MODE 1: (|c|+c)*norm1 -> out0 and out1.
template<int MODE>
__global__ __launch_bounds__(256) void wmma_gemm_kernel(
        const __bf16* __restrict__ A, const __bf16* __restrict__ Bt,
        float* __restrict__ out0, float* __restrict__ out1)
{
    // [buf][row][k], row stride 72 elems (144 B): 8 x b128 chunks + 16 B pad
    __shared__ __bf16 As[2][128][72];
    __shared__ __bf16 Bs[2][128][72];

    const int bm   = blockIdx.y, bn = blockIdx.x;
    const int tid  = threadIdx.x;
    const int wave = tid >> 5, lane = tid & 31;
    const int waveM = wave >> 1;       // 0..3 -> 32 rows each
    const int waveN = wave &  1;       // 0..1 -> 64 cols each
    const int half  = lane >> 4;
    const int l15   = lane & 15;

    v8f acc[2][4] = {};

    const __bf16* Ab = A  + (size_t)(bm * 128) * DNNGP;
    const __bf16* Bb = Bt + (size_t)(bn * 128) * DNNGP;

    // per-thread chunk assignment: 1024 b128 chunks per tile, 4 per thread
    const int cr = (tid * 4) >> 3;         // chunk row   (4 chunks contiguous)
    const int cc = (tid * 4) & 7;          // chunk col within row
    const unsigned int la_base =
        (unsigned int)(size_t)&As[0][cr][cc * 8];
    const unsigned int lb_base =
        (unsigned int)(size_t)&Bs[0][cr][cc * 8];
    const unsigned int buf_stride = (unsigned int)(128 * 72 * sizeof(__bf16));
    const __bf16* ga_base = Ab + (size_t)cr * DNNGP + cc * 8;
    const __bf16* gb_base = Bb + (size_t)cr * DNNGP + cc * 8;

    auto issue_tile = [&](int buf, int k0) {
        #pragma unroll
        for (int i = 0; i < 4; ++i) {      // 4 consecutive 16B chunks (64 B)
            async_copy_b128(la_base + buf * buf_stride + i * 16,
                            (unsigned long long)(ga_base + k0 + i * 8));
            async_copy_b128(lb_base + buf * buf_stride + i * 16,
                            (unsigned long long)(gb_base + k0 + i * 8));
        }
    };

    issue_tile(0, 0);
    int buf = 0;

    for (int k0 = 0; k0 < DNNGP; k0 += 64) {
        wait_async0();                      // this wave's copies landed
        __syncthreads();                    // everyone's copies landed
        if (k0 + 64 < DNNGP) issue_tile(buf ^ 1, k0 + 64);

        #pragma unroll
        for (int ks = 0; ks < 2; ++ks) {
            v16bf a[2], b[4];
            // A frag: lanes 0-15 K [0,8)+[16,24); lanes 16-31 K [8,16)+[24,32)
            #pragma unroll
            for (int mt = 0; mt < 2; ++mt) {
                int mrow = waveM * 32 + mt * 16 + l15;
                int c0   = ks * 32 + half * 8;
                v8bf lo = *(const v8bf*)&As[buf][mrow][c0];
                v8bf hi = *(const v8bf*)&As[buf][mrow][c0 + 16];
                a[mt] = cat8(lo, hi);
            }
            // B frag: lanes 0-15 K [0,8)+[16,24); lanes 16-31 K [8,16)+[24,32)
            #pragma unroll
            for (int nt = 0; nt < 4; ++nt) {
                int ncol = waveN * 64 + nt * 16 + l15;
                int kb   = ks * 32 + half * 8;
                v8bf lo = *(const v8bf*)&Bs[buf][ncol][kb];
                v8bf hi = *(const v8bf*)&Bs[buf][ncol][kb + 16];
                b[nt] = cat8(lo, hi);
            }
            #pragma unroll
            for (int mt = 0; mt < 2; ++mt)
                #pragma unroll
                for (int nt = 0; nt < 4; ++nt)
                    acc[mt][nt] = __builtin_amdgcn_wmma_f32_16x16x32_bf16(
                        false, a[mt], false, b[nt],
                        (short)0, acc[mt][nt], false, false);
        }
        __syncthreads();                    // reads done before buf rewritten
        buf ^= 1;
    }

    // epilogue: VGPR r -> lanes0-15 M=r, lanes16-31 M=8+r; N=lane%16
    #pragma unroll
    for (int mt = 0; mt < 2; ++mt)
        #pragma unroll
        for (int nt = 0; nt < 4; ++nt)
            #pragma unroll
            for (int r = 0; r < 8; ++r) {
                int row = bm * 128 + waveM * 32 + mt * 16 + half * 8 + r;
                int col = bn * 128 + waveN * 64 + nt * 16 + l15;
                float v = acc[mt][nt][r];
                if (MODE == 0) {
                    out0[(size_t)row * MDIM + col] = (v > 0.f) ? 0.03125f : 0.f;
                } else {
                    float z = (fabsf(v) + v) * 0.015625f;
                    out0[(size_t)row * MDIM + col] = z;
                    out1[(size_t)row * CATW + col] = z;
                }
            }
}

// ------------------------------------------------- count-sketch + FFT conv
__global__ __launch_bounds__(256) void fftconv_kernel(
        const float* __restrict__ z_ntk, const float* __restrict__ tmp,
        const float* __restrict__ sign1, const int* __restrict__ indx1,
        const float* __restrict__ sign2, const int* __restrict__ indx2,
        float* __restrict__ out)
{
    __shared__ float xr[MCS], xi[MCS], yr[MCS], yi[MCS];
    __shared__ float twr[MCS/2], twi[MCS/2];

    const int row = blockIdx.x;
    const int t   = threadIdx.x;
    const int T   = 256;
    const float TWO_PI = 6.28318530717958647692f;

    for (int i = t; i < MCS; i += T) { xr[i]=0.f; xi[i]=0.f; yr[i]=0.f; yi[i]=0.f; }
    for (int i = t; i < MCS/2; i += T) {
        float s, c;
        __sincosf(-TWO_PI * (float)i * (1.0f / MCS), &s, &c);
        twr[i] = c; twi[i] = s;
    }
    __syncthreads();

    const float* zr = z_ntk + (size_t)row * DNTK;
    for (int i = t; i < DNTK; i += T)
        atomicAdd(&xr[indx1[i]], zr[i] * sign1[i]);
    const float* tr = tmp + (size_t)row * MDIM;
    for (int i = t; i < MDIM; i += T)
        atomicAdd(&yr[indx2[i]], tr[i] * sign2[i]);
    __syncthreads();

    // forward DIF (natural -> bitrev) on X and Y
    for (int st = 11; st >= 0; --st) {
        const int s = 1 << st;
        for (int j = t; j < MCS/2; j += T) {
            int tt = j & (s - 1);
            int i0 = ((j >> st) << (st + 1)) + tt;
            int i1 = i0 + s;
            int j0 = tt << (11 - st);
            float wr = twr[j0], wi = twi[j0];
            {
                float ar = xr[i0], ai = xi[i0], br = xr[i1], bi = xi[i1];
                xr[i0] = ar + br; xi[i0] = ai + bi;
                float dr = ar - br, di = ai - bi;
                xr[i1] = dr*wr - di*wi; xi[i1] = dr*wi + di*wr;
            }
            {
                float ar = yr[i0], ai = yi[i0], br = yr[i1], bi = yi[i1];
                yr[i0] = ar + br; yi[i0] = ai + bi;
                float dr = ar - br, di = ai - bi;
                yr[i1] = dr*wr - di*wi; yi[i1] = dr*wi + di*wr;
            }
        }
        __syncthreads();
    }

    // pointwise complex multiply (bit-reversed order, aligned)
    for (int i = t; i < MCS; i += T) {
        float a = xr[i], b = xi[i], c = yr[i], d = yi[i];
        xr[i] = a*c - b*d; xi[i] = a*d + b*c;
    }
    __syncthreads();

    // inverse DIT (bitrev -> natural), conjugate twiddles
    for (int st = 0; st <= 11; ++st) {
        const int s = 1 << st;
        for (int j = t; j < MCS/2; j += T) {
            int tt = j & (s - 1);
            int i0 = ((j >> st) << (st + 1)) + tt;
            int i1 = i0 + s;
            int j0 = tt << (11 - st);
            float wr = twr[j0], wi = -twi[j0];
            float br = xr[i1]*wr - xi[i1]*wi;
            float bi = xr[i1]*wi + xi[i1]*wr;
            float ar = xr[i0], ai = xi[i0];
            xr[i0] = ar + br; xi[i0] = ai + bi;
            xr[i1] = ar - br; xi[i1] = ai - bi;
        }
        __syncthreads();
    }

    const size_t base = (size_t)NROWS * MDIM + (size_t)row * CATW + MDIM;
    const float scale = 1.0f / MCS;
    for (int i = t; i < MCS; i += T)
        out[base + i] = xr[i] * scale;
}

// ---------------------------------------------------------------- launcher
extern "C" void kernel_launch(void* const* d_in, const int* in_sizes, int n_in,
                              void* d_out, int out_size, void* d_ws, size_t ws_size,
                              hipStream_t stream) {
    (void)in_sizes; (void)n_in; (void)out_size; (void)ws_size;
    const float* z_nngp = (const float*)d_in[1];
    const float* z_ntk  = (const float*)d_in[2];
    const float* W0     = (const float*)d_in[3];
    const float* W1     = (const float*)d_in[4];
    const float* sign1  = (const float*)d_in[5];
    const int*   indx1  = (const int*)d_in[6];
    const float* sign2  = (const float*)d_in[7];
    const int*   indx2  = (const int*)d_in[8];

    float* out = (float*)d_out;

    // workspace layout (160 MB total)
    __bf16* Abf = (__bf16*)d_ws;                         // 64 MB
    __bf16* W0t = Abf + (size_t)NROWS * DNNGP;           // 32 MB
    __bf16* W1t = W0t + (size_t)MDIM  * DNNGP;           // 32 MB
    float*  tmp = (float*)(W1t + (size_t)MDIM * DNNGP);  // 32 MB

    // prep: convert A, transpose+convert W0/W1 (bf16 operands stay L2-resident)
    convert_bf16_kernel<<<(NROWS * DNNGP) / (256 * 4), 256, 0, stream>>>(z_nngp, Abf);
    dim3 tgrid(DNNGP / 64, MDIM / 64);                   // (128, 32)
    transpose_bf16_kernel<<<tgrid, 256, 0, stream>>>(W0, W0t);
    transpose_bf16_kernel<<<tgrid, 256, 0, stream>>>(W1, W1t);

    dim3 grid(MDIM / 128, NROWS / 128);                  // (16, 32)
    // tmp = step(z_nngp @ W0) * norm0
    wmma_gemm_kernel<0><<<grid, 256, 0, stream>>>(Abf, W0t, tmp, nullptr);
    // z_nngp_out = (|x|+x)*norm1 -> out[0..N*M1) and cat columns [0,2048)
    wmma_gemm_kernel<1><<<grid, 256, 0, stream>>>(Abf, W1t, out,
                                                  out + (size_t)NROWS * MDIM);
    // count-sketch + circular convolution -> cat columns [2048, 6144)
    fftconv_kernel<<<NROWS, 256, 0, stream>>>(z_ntk, tmp, sign1, indx1,
                                              sign2, indx2, out);
}
